// DA_conv1D_50036368998401
// MI455X (gfx1250) — matrix operand order repaired and checked by reading, not compiled, not run on hardware
//
#include <hip/hip_runtime.h>
#include <hip/hip_bf16.h>

#define B_  32
#define C_  64
#define L_  16384
#define K_  3
#define NL  128
#define TPB 256

typedef _Float16 v8h  __attribute__((ext_vector_type(8)));
typedef _Float16 v16h __attribute__((ext_vector_type(16)));
typedef float    v8f  __attribute__((ext_vector_type(8)));

__device__ __forceinline__ float lrelu(float x) { return x >= 0.f ? x : 0.1f * x; }

// ---------------------------------------------------------------------------
// Stage 1: tiny per-batch math -> kern[B][C][K] and att[B][C] in workspace.
// ---------------------------------------------------------------------------
__global__ __launch_bounds__(TPB) void precompute_kernel(
    const float* __restrict__ x1,    // [B,64]
    const float* __restrict__ W1,    // [64,64]
    const float* __restrict__ W2,    // [192,64]
    const float* __restrict__ ca_w1, // [8,64]
    const float* __restrict__ ca_w2, // [64,8]
    float* __restrict__ kern_out,    // [B*C*K]
    float* __restrict__ att_out)     // [B*C]
{
    __shared__ float hs[B_][64];
    __shared__ float gs[B_][8];
    const int tid = threadIdx.x;

    // h = lrelu(x1 @ W1^T)
    for (int i = tid; i < B_ * 64; i += TPB) {
        int b = i >> 6, j = i & 63;
        float s = 0.f;
        #pragma unroll 8
        for (int q = 0; q < 64; ++q) s += x1[b * 64 + q] * W1[j * 64 + q];
        hs[b][j] = lrelu(s);
    }
    // g = lrelu(x1 @ ca_w1^T)
    for (int i = tid; i < B_ * 8; i += TPB) {
        int b = i >> 3, r = i & 7;
        float s = 0.f;
        #pragma unroll 8
        for (int q = 0; q < 64; ++q) s += x1[b * 64 + q] * ca_w1[r * 64 + q];
        gs[b][r] = lrelu(s);
    }
    __syncthreads();
    // kern = h @ W2^T  (reshaped [B,C,K])
    for (int i = tid; i < B_ * C_ * K_; i += TPB) {
        int b = i / (C_ * K_), ck = i % (C_ * K_);
        float s = 0.f;
        #pragma unroll 8
        for (int q = 0; q < 64; ++q) s += hs[b][q] * W2[ck * 64 + q];
        kern_out[i] = s;
    }
    // att = sigmoid(g @ ca_w2^T)
    for (int i = tid; i < B_ * C_; i += TPB) {
        int b = i >> 6, c = i & 63;
        float s = 0.f;
        #pragma unroll
        for (int r = 0; r < 8; ++r) s += gs[b][r] * ca_w2[c * 8 + r];
        att_out[i] = 1.f / (1.f + __expf(-s));
    }
}

// ---------------------------------------------------------------------------
// Stage 2: fused depthwise conv + lrelu + 1x1 conv (WMMA) + bias + x0*att.
// One workgroup = one (b, 128-wide L tile). 8 waves, each owns 16 columns.
// x0 tile is staged to LDS with GLOBAL_LOAD_ASYNC_TO_LDS_B128 (ASYNCcnt).
// ---------------------------------------------------------------------------
__global__ __launch_bounds__(TPB) void fused_main_kernel(
    const float* __restrict__ x0,     // [B,C,L]
    const float* __restrict__ conv_w, // [C,C]  (out,in)
    const float* __restrict__ conv_b, // [C]
    const float* __restrict__ kern,   // [B,C,K]
    const float* __restrict__ att,    // [B,C]
    float* __restrict__ out)          // [B,C,L]
{
    // xs rows: [3]=left halo, body [4..131] (16B-aligned), [132]=right halo.
    // Row stride 136 floats = 544B: 16B-aligned rows, 8-bank skew per row.
    __shared__ __align__(16) float    xs[C_][NL + 8];
    __shared__ __align__(16) _Float16 ts[C_][NL + 8];   // stride 272B rows
    __shared__ __align__(16) _Float16 wsm[C_][72];      // conv_w f16, 144B rows
    __shared__ float kern_s[C_][K_];
    __shared__ float att_s[C_];
    __shared__ float bias_s[C_];

    const int tid       = threadIdx.x;
    const int tilesPerB = L_ / NL;                      // 128
    const int b         = blockIdx.x / tilesPerB;
    const int L0        = (blockIdx.x % tilesPerB) * NL;
    const float* x0b    = x0 + (size_t)b * C_ * L_;

    // ---- async DMA of the x0 tile body into LDS (16B per lane per issue) --
    // Each wave iteration covers one channel row: 32 lanes x 16B = 512B,
    // contiguous in both global memory and LDS.
    for (int i = tid; i < C_ * (NL / 4); i += TPB) {
        int c = i >> 5, v = i & 31;                     // NL/4 == 32 chunks/row
        const float* gsrc = x0b + (size_t)c * L_ + L0 + v * 4;
        unsigned ldst = (unsigned)(uintptr_t)(void*)&xs[c][4 + v * 4];
        asm volatile("global_load_async_to_lds_b128 %0, %1, off"
                     :: "v"(ldst), "v"(gsrc) : "memory");
    }

    // ---- overlap with the async copy: conv_w -> f16 LDS, scalars ----------
    for (int i = tid; i < C_ * C_; i += TPB)
        wsm[i >> 6][i & 63] = (_Float16)conv_w[i];

    if (tid < C_) {
        bias_s[tid]    = conv_b[tid];
        att_s[tid]     = att[b * C_ + tid];
        kern_s[tid][0] = kern[(b * C_ + tid) * K_ + 0];
        kern_s[tid][1] = kern[(b * C_ + tid) * K_ + 1];
        kern_s[tid][2] = kern[(b * C_ + tid) * K_ + 2];
    } else if (tid < 2 * C_) {          // left halo (zero-pad at L==0)
        int c = tid - C_;
        xs[c][3] = (L0 > 0) ? x0b[(size_t)c * L_ + L0 - 1] : 0.f;
    } else if (tid < 3 * C_) {          // right halo (zero-pad at L end)
        int c = tid - 2 * C_;
        xs[c][NL + 4] = (L0 + NL < L_) ? x0b[(size_t)c * L_ + L0 + NL] : 0.f;
    }

    // All async LDS writes from this wave must land before the barrier.
    asm volatile("s_wait_asynccnt 0" ::: "memory");
    __syncthreads();

    // ---- depthwise 3-tap (cross-correlation) + lrelu -> f16 tile ----------
    for (int i = tid; i < C_ * NL; i += TPB) {
        int c = i >> 7, l = i & (NL - 1);
        float v = xs[c][l + 3] * kern_s[c][0]
                + xs[c][l + 4] * kern_s[c][1]
                + xs[c][l + 5] * kern_s[c][2];
        ts[c][l] = (_Float16)lrelu(v);
    }
    __syncthreads();

    // ---------------- WMMA GEMM: D[o,n] = sum_c conv_w[o,c] * ts[c,n] -----
    const int lane  = tid & 31;
    const int wv    = tid >> 5;       // 0..7 -> N tile
    const int n0    = wv * 16;
    const int mlane = lane & 15;      // A: lane -> M row ; D: lane -> N col
    const int kg    = lane >> 4;      // half-wave selector

    v8f acc[4];
    #pragma unroll
    for (int mt = 0; mt < 4; ++mt) acc[mt] = (v8f){};

    #pragma unroll
    for (int ks = 0; ks < 2; ++ks) {
        // B fragment (32x16, K x N): lane n holds K = kg*8..+7 and 16+kg*8..+7.
        int cb = ks * 32 + kg * 8;
        v16h bf;
        #pragma unroll
        for (int q = 0; q < 8; ++q) { bf[q] = ts[cb + q][n0 + mlane]; bf[q + 8] = ts[cb + 16 + q][n0 + mlane]; }

        #pragma unroll
        for (int mt = 0; mt < 4; ++mt) {
            // A fragment (16x32, M x K): lane m holds K = kg*8..+7 (VGPR0-3)
            // and K = 16+kg*8..+7 (VGPR4-7), per ISA 16-bit A layout.
            int m  = mt * 16 + mlane;
            int ka = ks * 32 + kg * 8;
            v8h alo = *(const v8h*)&wsm[m][ka];
            v8h ahi = *(const v8h*)&wsm[m][ka + 16];
            v16h af;
            #pragma unroll
            for (int q = 0; q < 8; ++q) { af[q] = alo[q]; af[q + 8] = ahi[q]; }

            acc[mt] = __builtin_amdgcn_wmma_f32_16x16x32_f16(
                /*neg_a=*/false, af, /*neg_b=*/false, bf,
                /*c_mod=*/(short)0, acc[mt],
                /*reuse_a=*/false, /*reuse_b=*/false);
        }
    }

    // ---------------- epilogue: + bias + x0*att, coalesced f32 stores ------
    const int n     = n0 + mlane;     // D layout: N = lane&15
    const int lglob = L0 + n;
    #pragma unroll
    for (int mt = 0; mt < 4; ++mt) {
        #pragma unroll
        for (int r = 0; r < 8; ++r) {
            int m = mt * 16 + kg * 8 + r;   // D: VGPR r -> M=r (lo) / 8+r (hi)
            float v = acc[mt][r] + bias_s[m] + xs[m][n + 4] * att_s[m];
            out[((size_t)b * C_ + m) * L_ + lglob] = v;
        }
    }
}

// ---------------------------------------------------------------------------
extern "C" void kernel_launch(void* const* d_in, const int* in_sizes, int n_in,
                              void* d_out, int out_size, void* d_ws, size_t ws_size,
                              hipStream_t stream) {
    (void)in_sizes; (void)n_in; (void)out_size; (void)ws_size;
    const float* x0     = (const float*)d_in[0];
    const float* x1     = (const float*)d_in[1];
    const float* W1     = (const float*)d_in[2];
    const float* W2     = (const float*)d_in[3];
    const float* conv_w = (const float*)d_in[4];
    const float* conv_b = (const float*)d_in[5];
    const float* ca_w1  = (const float*)d_in[6];
    const float* ca_w2  = (const float*)d_in[7];
    float* out  = (float*)d_out;
    float* kern = (float*)d_ws;                 // B*C*K = 6144 floats
    float* att  = kern + B_ * C_ * K_;          // B*C   = 2048 floats

    precompute_kernel<<<1, TPB, 0, stream>>>(x1, W1, W2, ca_w1, ca_w2, kern, att);

    const int nblocks = B_ * (L_ / NL);         // 4096 workgroups
    fused_main_kernel<<<nblocks, TPB, 0, stream>>>(x0, conv_w, conv_b, kern, att, out);
}
